// MultiHeadAttentionVanilla_309237646005
// MI455X (gfx1250) — matrix operand order repaired and checked by reading, not compiled, not run on hardware
//
#include <hip/hip_runtime.h>

// Problem constants (B=2, S=2048, E=768, H=12, D=64)
#define SLEN 2048
#define NB   2
#define NH   12
#define EMB  768
#define HD   64

typedef __attribute__((ext_vector_type(16))) __bf16 v16bf;
typedef __attribute__((ext_vector_type(8)))  float  v8f;
typedef __attribute__((ext_vector_type(4)))  unsigned int v4u;
typedef __attribute__((ext_vector_type(8)))  int v8i;
typedef __attribute__((ext_vector_type(4)))  int v4i;

union ABu { unsigned int u[8]; v16bf v; };

__device__ __forceinline__ unsigned short f2bf(float f) {
    unsigned int u = __builtin_bit_cast(unsigned int, f);
    u += 0x7FFFu + ((u >> 16) & 1u);   // round-to-nearest-even
    return (unsigned short)(u >> 16);
}

__device__ __forceinline__ v8f wmma_bf16(const ABu& a, const ABu& b, v8f c) {
    return __builtin_amdgcn_wmma_f32_16x16x32_bf16(
        /*neg_a=*/false, a.v, /*neg_b=*/false, b.v,
        /*c_mod=*/(short)0, c, /*reuse_a=*/false, /*reuse_b=*/false);
}

// LDS byte offset: flat LDS-aperture pointers carry the DS offset in addr[31:0]
__device__ __forceinline__ unsigned lds_off(const void* p) {
    return (unsigned)(unsigned long long)p;
}

// Async 16B memory->LDS copy (per-lane), tracked by ASYNCcnt
__device__ __forceinline__ void async_copy16(unsigned lds_byte, const void* gaddr) {
    asm volatile("global_load_async_to_lds_b128 %0, %1, off"
                 :: "v"(lds_byte), "v"(gaddr) : "memory");
}
__device__ __forceinline__ void wait_async0() {
    asm volatile("s_wait_asynccnt 0" ::: "memory");
}

// ---------------------------------------------------------------------------
// TDM: DMA one 2D bf16 tile [rows x 64 elems] (row stride 64 elems in memory)
// into LDS with 16B padding after every 128B -> padded LDS row stride 144B.
// D# layout per CDNA5 ISA 8.3/8.4 (group0: count/lds/global/type,
// group1: data_size, pad, dims, strides). Groups 2/3/4 zero (2-D tensor).
// This toolchain exposes the 6-arg builtin: (v4u, v8i, v4i, v4i, v8i, i32).
// ---------------------------------------------------------------------------
__device__ __forceinline__ void tdm_load_tile_64xD(unsigned lds_byte,
                                                   const void* gaddr,
                                                   unsigned rows) {
    unsigned long long ga = (unsigned long long)gaddr;
    v4u g0;
    g0.x = 1u;                                      // count=1, user descriptor
    g0.y = lds_byte;                                // lds_addr (bytes)
    g0.z = (unsigned)(ga & 0xFFFFFFFFu);            // global_addr[31:0]
    g0.w = (unsigned)((ga >> 32) & 0x01FFFFFFu)     // global_addr[56:32]
         | (2u << 30);                              // type=2 ("image")
    v8i g1;
    g1[0] = (1 << 16)    // data_size = 2 bytes
          | (1 << 20)    // pad_enable
          | (4 << 22)    // pad_interval: pad after 32 dwords (128B)
          | (3 << 25);   // pad_amount: 4 dwords (16B)
    g1[1] = (int)(64u << 16);     // tensor_dim0 = 64 (bits 79:48, low16 here)
    g1[2] = (int)(2048u << 16);   // tensor_dim0 hi = 0; tensor_dim1 = 2048
    g1[3] = (int)(64u << 16);     // tensor_dim1 hi = 0; tile_dim0 = 64
    g1[4] = (int)rows;            // tile_dim1 = rows; tile_dim2 = 0
    g1[5] = 64;                   // tensor_dim0_stride = 64 elements
    g1[6] = 0;                    // stride hi / tensor_dim1_stride lo (unused 2D)
    g1[7] = 0;
    v4i g2 = {0, 0, 0, 0};
    v4i g3 = {0, 0, 0, 0};
    v8i g4 = {0, 0, 0, 0, 0, 0, 0, 0};
    __builtin_amdgcn_tensor_load_to_lds(g0, g1, g2, g3, g4, 0);
}
__device__ __forceinline__ void wait_tensor0() {
    __builtin_amdgcn_s_wait_tensorcnt(0);
}

// ---------------------------------------------------------------------------
// fp32 -> bf16 conversion, 4 elements per thread (grid-stride)
// ---------------------------------------------------------------------------
__global__ void cvt_bf16_kernel(const float* __restrict__ in,
                                unsigned short* __restrict__ out, int n4) {
    for (int i = blockIdx.x * blockDim.x + threadIdx.x; i < n4;
         i += gridDim.x * blockDim.x) {
        float4 f = ((const float4*)in)[i];
        unsigned short r[4] = { f2bf(f.x), f2bf(f.y), f2bf(f.z), f2bf(f.w) };
        ((uint2*)out)[i] = *(const uint2*)r;
    }
}

// ---------------------------------------------------------------------------
// GEMM: Y[m,n] = sum_k X[m,k] * W[n,k] + bias[n]   (X:[Mtot,768] W:[768,768], bf16)
// MODE 0: store fp32 row-major [Mtot,768]            (final projection)
// MODE 1: RoPE then store bf16 [B,H,S,D]             (Q, K)
// MODE 2: store bf16 [B,H,S,D]                       (V)
// 256 threads = 8 waves; tile 128(M) x 64(N); K-step 32.
// Double-buffered LDS, async-to-LDS staging overlapped with WMMA compute.
// ---------------------------------------------------------------------------
#define XS_STRIDE 20   // dwords per LDS row (16 valid + pad; 80B => 16B aligned)
#define WS_STRIDE 20

template <int MODE>
__global__ __launch_bounds__(256)
void gemm_kernel(const unsigned short* __restrict__ X,
                 const unsigned short* __restrict__ W,
                 const float* __restrict__ bias,
                 void* __restrict__ outp) {
    __shared__ unsigned int Xs[2][128 * XS_STRIDE];
    __shared__ unsigned int Ws[2][64 * WS_STRIDE];

    const int t    = threadIdx.x;
    const int lane = t & 31;
    const int wv   = t >> 5;            // wave 0..7
    const int hi   = lane >> 4;         // half-wave 0/1
    const int m0   = blockIdx.x * 128;
    const int n0   = blockIdx.y * 64;

    const unsigned int* Xd = (const unsigned int*)X;   // row stride 384 dwords
    const unsigned int* Wd = (const unsigned int*)W;

    const unsigned xs_base = lds_off(&Xs[0][0]);
    const unsigned ws_base = lds_off(&Ws[0][0]);

    // issue async staging of one 32-wide K tile into buffer `buf`
    auto stage = [&](int buf, int k0) {
        const int base_dw = k0 >> 1;
        for (int i = 0; i < 2; ++i) {
            int idx = t + i * 256;
            int row = idx >> 2, c4 = idx & 3;
            async_copy16(xs_base + (unsigned)(buf * 128 * XS_STRIDE +
                                              row * XS_STRIDE + c4 * 4) * 4u,
                         Xd + (m0 + row) * 384 + base_dw + c4 * 4);
        }
        int row = t >> 2, c4 = t & 3;
        async_copy16(ws_base + (unsigned)(buf * 64 * WS_STRIDE +
                                          row * WS_STRIDE + c4 * 4) * 4u,
                     Wd + (n0 + row) * 384 + base_dw + c4 * 4);
    };

    v8f z = {};
    v8f acc[4] = { z, z, z, z };

    stage(0, 0);
    wait_async0();
    __syncthreads();

    for (int k0 = 0; k0 < EMB; k0 += 32) {
        const int cur = (k0 >> 5) & 1;
        if (k0 + 32 < EMB) stage(cur ^ 1, k0 + 32);   // overlap fetch w/ compute

        const unsigned int* Xc = &Xs[cur][0];
        const unsigned int* Wc = &Ws[cur][0];

        // A operand: this wave's 16 rows, K-pairs in contiguous order (k = 16*hi + s)
        ABu a;
        const int lrow = wv * 16 + (lane & 15);
        for (int v = 0; v < 8; ++v) {
            int dw = (hi ? 8 : 0) + v;
            a.u[v] = Xc[lrow * XS_STRIDE + dw];
        }
        for (int nt = 0; nt < 4; ++nt) {
            ABu bm;
            const int nl = nt * 16 + (lane & 15);
            for (int v = 0; v < 8; ++v)
                bm.u[v] = Wc[nl * WS_STRIDE + (hi ? 8 : 0) + v];
            acc[nt] = wmma_bf16(a, bm, acc[nt]);
        }

        wait_async0();     // next tile landed in LDS
        __syncthreads();   // all waves done reading `cur`
    }

    // epilogue
    for (int nt = 0; nt < 4; ++nt) {
        const int n  = n0 + nt * 16 + (lane & 15);
        const float bn = bias[n];
        float vals[8];
        for (int r = 0; r < 8; ++r) vals[r] = acc[nt][r] + bn;

        if (MODE == 0) {
            float* out = (float*)outp;
            for (int r = 0; r < 8; ++r) {
                int m = m0 + wv * 16 + r + 8 * hi;
                out[m * EMB + n] = vals[r];
            }
        } else {
            unsigned short* out = (unsigned short*)outp;
            const int h = n / HD, d = n % HD, par = n & 1;
            // inv_freq = 10000^(-(2*(d/2))/64)
            const float invf = __expf((float)(d >> 1) * (-2.0f / 64.0f) * 9.2103403720f);
            for (int r = 0; r < 8; ++r) {
                int m = m0 + wv * 16 + r + 8 * hi;
                int s = m & (SLEN - 1), b = m >> 11;
                float y = vals[r];
                if (MODE == 1) {
                    float partner = __shfl_xor(vals[r], 1, 32);  // pair element d^1
                    float ang = (float)s * invf;
                    float sn, cs;
                    __sincosf(ang, &sn, &cs);
                    y = par ? (vals[r] * cs + partner * sn)
                            : (vals[r] * cs - partner * sn);
                }
                out[((b * NH + h) * SLEN + s) * HD + d] = f2bf(y);
            }
        }
    }
}

// ---------------------------------------------------------------------------
// Flash attention: block = (64 q-rows, one (b,h)); 128 threads = 4 waves.
// Streams keys in blocks of 64 with online softmax; bf16 WMMA for QK^T and PV.
// K tile DMA'd by the Tensor Data Mover (with LDS row padding); V tile
// transposed manually during staging while the TDM runs.
// ---------------------------------------------------------------------------
#define KT_STRIDE 36   // dwords per LDS row for K/V tiles (144B, 16B aligned)

__global__ __launch_bounds__(128)
void attn_kernel(const unsigned short* __restrict__ Qr,
                 const unsigned short* __restrict__ Kr,
                 const unsigned short* __restrict__ Vr,
                 const float* __restrict__ rel,
                 const unsigned char* __restrict__ maskp,
                 const float* __restrict__ coeffp,
                 unsigned short* __restrict__ attnbuf) {
    __shared__ unsigned int Kt[64 * KT_STRIDE];   // [key][d]  row-major (TDM-filled)
    __shared__ unsigned int Vt[64 * KT_STRIDE];   // [d][key]  transposed
    __shared__ unsigned int Ps[4 * 16 * 32];      // per-wave P scratch [16][64] bf16

    const int t    = threadIdx.x;
    const int lane = t & 31;
    const int wv   = t >> 5;
    const int hi   = lane >> 4;
    const int q0   = blockIdx.x * 64;
    const int bh   = blockIdx.y;
    const int b    = bh / NH, h = bh % NH;
    const float coeff = coeffp[h];
    const float scale = 0.125f;  // 1/sqrt(64)

    const unsigned int* Qd = (const unsigned int*)Qr;
    const unsigned int* Kd = (const unsigned int*)Kr + (size_t)bh * SLEN * 32;
    const unsigned short* Vp = Vr + (size_t)bh * SLEN * HD;
    const unsigned kt_base = lds_off(&Kt[0]);

    // Q A-operands for this wave's 16 rows (two K-steps of 32 over d)
    ABu aq[2];
    const int qrow = q0 + wv * 16 + (lane & 15);
    for (int ks = 0; ks < 2; ++ks)
        for (int v = 0; v < 8; ++v) {
            int d = ks * 32 + (hi ? 16 : 0) + v * 2;
            aq[ks].u[v] = Qd[((size_t)bh * SLEN + qrow) * 32 + (d >> 1)];
        }
    float pbq[8];
    for (int r = 0; r < 8; ++r)
        pbq[r] = rel[b * SLEN + q0 + wv * 16 + r + 8 * hi];

    float mrow[8], lsum[8];
    for (int r = 0; r < 8; ++r) { mrow[r] = -1e30f; lsum[r] = 0.0f; }
    v8f z = {};
    v8f Oacc[4] = { z, z, z, z };

    for (int kk = 0; kk < SLEN; kk += 64) {
        __syncthreads();
        // K tile: one TDM descriptor issued by wave 0 (64 keys x 128B,
        // padded to 144B rows in LDS); runs while we stage V manually.
        if (wv == 0)
            tdm_load_tile_64xD(kt_base, Kd + (size_t)kk * 32, 64);
        // stage V tile transposed (manual scatter; TDM fills Kt meanwhile)
        for (int i = 0; i < 4; ++i) {
            int idx = t + i * 128;
            int key = idx >> 3, d0 = (idx & 7) * 8;
            uint4 vv = *(const uint4*)(Vp + (size_t)(kk + key) * HD + d0);
            const unsigned short* pv = (const unsigned short*)&vv;
            unsigned short* vt16 = (unsigned short*)Vt;
            for (int j = 0; j < 8; ++j)
                vt16[(d0 + j) * (KT_STRIDE * 2) + key] = pv[j];
        }
        wait_tensor0();    // wave 0: K tile landed (no-op for other waves)
        __syncthreads();

        // ---- scores S = Q K^T * scale + coeff*(pbq - pbk), masked ----
        float sv[4][8], bm[8], pbk[4], pmask[4];
        for (int r = 0; r < 8; ++r) bm[r] = -1e30f;
        for (int nt = 0; nt < 4; ++nt) {
            int key = kk + nt * 16 + (lane & 15);
            pbk[nt]   = rel[b * SLEN + key];
            pmask[nt] = maskp[b * SLEN + key] ? 1.0f : 0.0f;
            v8f c = z;
            for (int ks = 0; ks < 2; ++ks) {
                ABu bmat;
                int kl = nt * 16 + (lane & 15);
                for (int v = 0; v < 8; ++v)
                    bmat.u[v] = Kt[kl * KT_STRIDE + ks * 16 + (hi ? 8 : 0) + v];
                c = wmma_bf16(aq[ks], bmat, c);
            }
            for (int r = 0; r < 8; ++r) {
                float s = c[r] * scale + coeff * (pbq[r] - pbk[nt]);
                s = (pmask[nt] > 0.0f) ? s : -1e30f;
                sv[nt][r] = s;
                bm[r] = fmaxf(bm[r], s);
            }
        }
        // row-max across the 16 lanes holding each row
        for (int off = 1; off < 16; off <<= 1)
            for (int r = 0; r < 8; ++r)
                bm[r] = fmaxf(bm[r], __shfl_xor(bm[r], off, 32));

        float alpha[8];
        for (int r = 0; r < 8; ++r) {
            float mn = fmaxf(mrow[r], bm[r]);
            alpha[r] = __expf(mrow[r] - mn);
            mrow[r]  = mn;
        }

        // ---- P = exp(S - m), write to LDS (C-layout -> A-layout transpose) ----
        unsigned short* Pw = (unsigned short*)(Ps + wv * 16 * 32);
        float rs[8];
        for (int r = 0; r < 8; ++r) rs[r] = 0.0f;
        for (int nt = 0; nt < 4; ++nt)
            for (int r = 0; r < 8; ++r) {
                float p = (pmask[nt] > 0.0f) ? __expf(sv[nt][r] - mrow[r]) : 0.0f;
                rs[r] += p;
                Pw[(r + 8 * hi) * 64 + nt * 16 + (lane & 15)] = f2bf(p);
            }
        for (int off = 1; off < 16; off <<= 1)
            for (int r = 0; r < 8; ++r) rs[r] += __shfl_xor(rs[r], off, 32);
        for (int r = 0; r < 8; ++r) lsum[r] = lsum[r] * alpha[r] + rs[r];

        __syncthreads();  // order P stores before A-layout reads

        // rescale O then accumulate P @ V
        for (int dt = 0; dt < 4; ++dt)
            for (int r = 0; r < 8; ++r)
                Oacc[dt][r] = Oacc[dt][r] * alpha[r];

        ABu ap[2];
        const unsigned int* Pr = Ps + wv * 16 * 32;
        for (int ks = 0; ks < 2; ++ks)
            for (int v = 0; v < 8; ++v)
                ap[ks].u[v] = Pr[(lane & 15) * 32 + ks * 16 +
                                 (hi ? 8 : 0) + v];
        for (int dt = 0; dt < 4; ++dt)
            for (int ks = 0; ks < 2; ++ks) {
                ABu bmat;
                int dl = dt * 16 + (lane & 15);
                for (int v = 0; v < 8; ++v)
                    bmat.u[v] = Vt[dl * KT_STRIDE + ks * 16 + (hi ? 8 : 0) + v];
                Oacc[dt] = wmma_bf16(ap[ks], bmat, Oacc[dt]);
            }
    }

    // normalize and store to attnbuf [B,S,E] bf16
    for (int r = 0; r < 8; ++r)
        lsum[r] = (lsum[r] > 0.0f) ? 1.0f / lsum[r] : 0.0f;
    for (int dt = 0; dt < 4; ++dt) {
        int d = dt * 16 + (lane & 15);
        for (int r = 0; r < 8; ++r) {
            int q = q0 + wv * 16 + r + 8 * hi;
            attnbuf[((size_t)(b * SLEN + q)) * EMB + h * HD + d] =
                f2bf(Oacc[dt][r] * lsum[r]);
        }
    }
}

// ---------------------------------------------------------------------------
// Launcher
// ---------------------------------------------------------------------------
extern "C" void kernel_launch(void* const* d_in, const int* in_sizes, int n_in,
                              void* d_out, int out_size, void* d_ws, size_t ws_size,
                              hipStream_t stream) {
    (void)in_sizes; (void)n_in; (void)out_size; (void)ws_size;
    const float* query = (const float*)d_in[0];
    const float* keyi  = (const float*)d_in[1];
    const float* value = (const float*)d_in[2];
    const float* rel   = (const float*)d_in[3];
    const unsigned char* mask = (const unsigned char*)d_in[4];
    const float* wq_w = (const float*)d_in[5];
    const float* wq_b = (const float*)d_in[6];
    const float* wk_w = (const float*)d_in[7];
    const float* wk_b = (const float*)d_in[8];
    const float* wv_w = (const float*)d_in[9];
    const float* wv_b = (const float*)d_in[10];
    const float* fc_w = (const float*)d_in[11];
    const float* fc_b = (const float*)d_in[12];
    const float* coeff = (const float*)d_in[13];

    const int XE = NB * SLEN * EMB;   // 3,145,728
    const int WE = EMB * EMB;         //   589,824

    char* ws = (char*)d_ws;
    size_t off = 0;
    auto alloc = [&](size_t bytes) -> void* {
        void* p = ws + off;
        off = (off + bytes + 255) & ~(size_t)255;
        return p;
    };
    unsigned short* Xq  = (unsigned short*)alloc((size_t)XE * 2);
    unsigned short* Xk  = (unsigned short*)alloc((size_t)XE * 2);
    unsigned short* Xv  = (unsigned short*)alloc((size_t)XE * 2);
    unsigned short* Wq  = (unsigned short*)alloc((size_t)WE * 2);
    unsigned short* Wk  = (unsigned short*)alloc((size_t)WE * 2);
    unsigned short* Wv  = (unsigned short*)alloc((size_t)WE * 2);
    unsigned short* Wfc = (unsigned short*)alloc((size_t)WE * 2);
    unsigned short* Qr  = (unsigned short*)alloc((size_t)XE * 2);  // [B,H,S,D]
    unsigned short* Kr  = (unsigned short*)alloc((size_t)XE * 2);
    unsigned short* Vr  = (unsigned short*)alloc((size_t)XE * 2);
    unsigned short* Abf = (unsigned short*)alloc((size_t)XE * 2);  // [B,S,E]

    dim3 cblk(256), cgrd(512);
    cvt_bf16_kernel<<<cgrd, cblk, 0, stream>>>(query, Xq, XE / 4);
    cvt_bf16_kernel<<<cgrd, cblk, 0, stream>>>(keyi,  Xk, XE / 4);
    cvt_bf16_kernel<<<cgrd, cblk, 0, stream>>>(value, Xv, XE / 4);
    cvt_bf16_kernel<<<cgrd, cblk, 0, stream>>>(wq_w, Wq,  WE / 4);
    cvt_bf16_kernel<<<cgrd, cblk, 0, stream>>>(wk_w, Wk,  WE / 4);
    cvt_bf16_kernel<<<cgrd, cblk, 0, stream>>>(wv_w, Wv,  WE / 4);
    cvt_bf16_kernel<<<cgrd, cblk, 0, stream>>>(fc_w, Wfc, WE / 4);

    dim3 gblk(256), ggrd(NB * SLEN / 128, EMB / 64);   // 32 x 12
    gemm_kernel<1><<<ggrd, gblk, 0, stream>>>(Xq, Wq, wq_b, Qr);  // Q + RoPE
    gemm_kernel<1><<<ggrd, gblk, 0, stream>>>(Xk, Wk, wk_b, Kr);  // K + RoPE
    gemm_kernel<2><<<ggrd, gblk, 0, stream>>>(Xv, Wv, wv_b, Vr);  // V

    dim3 ablk(128), agrd(SLEN / 64, NB * NH);          // 32 x 24
    attn_kernel<<<agrd, ablk, 0, stream>>>(Qr, Kr, Vr, rel, mask, coeff, Abf);

    gemm_kernel<0><<<ggrd, gblk, 0, stream>>>(Abf, Wfc, fc_b, (float*)d_out);
}
